// MattNet_65627100283272
// MI455X (gfx1250) — matrix operand order repaired and checked by reading, not compiled, not run on hardware
//
#include <hip/hip_runtime.h>
#include <hip/hip_bf16.h>
#include <math.h>
#include <stdint.h>

typedef __bf16 bf16_t;
typedef __attribute__((ext_vector_type(16))) __bf16 v16bf;
typedef __attribute__((ext_vector_type(8)))  __bf16 v8bf;
typedef __attribute__((ext_vector_type(8)))  float  v8f;

#define DI __device__ __forceinline__

DI float sigm(float x) { return 1.f / (1.f + expf(-x)); }

// CDNA5 async copy: LDS[lds_dst] = MEM[gsrc], 16 bytes per lane, ASYNCcnt.
// (CDNA5 ISA GLOBAL_LOAD_ASYNC_TO_LDS_B128, opcode 98; VDST holds
// the per-lane LDS byte address, VADDR the 64-bit global address.)
DI void async_stage_b128(void* lds_dst, const void* gsrc) {
  unsigned lds_addr = (unsigned)(size_t)lds_dst;  // LDS aperture: low 32 bits = LDS offset
  asm volatile("global_load_async_to_lds_b128 %0, %1, off"
               :: "v"(lds_addr), "v"(gsrc)
               : "memory");
}
DI void wait_async0() {
#if __has_builtin(__builtin_amdgcn_s_wait_asynccnt)
  __builtin_amdgcn_s_wait_asynccnt(0);
#else
  asm volatile("s_wait_asynccnt 0x0" ::: "memory");
#endif
}

// A-matrix fragment for V_WMMA_F32_16X16X32_BF16 from row-major A[M,K] (bf16),
// loaded from global memory. Caller passes Ap = A + row*K + half*16.
// lane half 0: K = k+0..7 (VGPR0..3), k+8..15 (VGPR4..7)
// lane half 1: K = k+16..23,         k+24..31
DI v16bf load_a_frag(const bf16_t* Ap, int k) {
  v8bf lo = *(const v8bf*)(Ap + k);
  v8bf hi = *(const v8bf*)(Ap + k + 8);
  v16bf f;
#pragma unroll
  for (int i = 0; i < 8; ++i) { f[i] = lo[i]; f[i + 8] = hi[i]; }
  return f;
}

// Same fragment gather from an LDS row (32 K-values, row base pointer).
DI v16bf frag_lds(const bf16_t* rp, int half) {
  v8bf lo = *(const v8bf*)(rp + half * 16);
  v8bf hi = *(const v8bf*)(rp + half * 16 + 8);
  v16bf f;
#pragma unroll
  for (int i = 0; i < 8; ++i) { f[i] = lo[i]; f[i + 8] = hi[i]; }
  return f;
}

DI float apply_act(float x, int act) {
  if (act == 1) return fmaxf(x, 0.f);
  if (act == 2) return x > 0.f ? x : 0.01f * x;
  return x;
}

DI void store_tile(float* __restrict__ C, bf16_t* __restrict__ Cb, int N,
                   int rbase, int col, v8f acc, float badd, int act) {
#pragma unroll
  for (int v = 0; v < 8; ++v) {
    float x = apply_act(acc[v] + badd, act);
    size_t idx = (size_t)(rbase + v) * N + col;
    C[idx] = x;
    if (Cb) Cb[idx] = (bf16_t)x;
  }
}

// ---------------------------------------------------------------------------
// GEMM: C[M,N] = act(A[M,K] @ W[N,K]^T + bias1 + bias2); A,W bf16; C f32.
// Block = 128 threads (4 waves). Block tile 32(M) x 128(N); each wave owns a
// 2Mx2N quad of 16x16 tiles (4 WMMAs sharing 2 A-frags + 2 B-frags).
// The 32x32 A panel is staged via double-buffered LDS using CDNA5
// global_load_async_to_lds_b128 (ASYNCcnt), overlapping the WMMA stream.
// All waves run an identical EXEC-stable loop; only the epilogue is guarded.
// Requires: N % 32 == 0, K % 32 == 0, grid.y = Mpad/32 (M padded to 32).
// ---------------------------------------------------------------------------
__global__ __launch_bounds__(128)
void gemm_bf16_wmma(const bf16_t* __restrict__ A, const bf16_t* __restrict__ W,
                    const float* __restrict__ bias1, const float* __restrict__ bias2,
                    float* __restrict__ C, bf16_t* __restrict__ Cb,
                    int N, int K, int act)
{
  __shared__ bf16_t As[2][32][40];           // row stride 80B (16B aligned)
  const int lane = threadIdx.x & 31;
  const int wave = threadIdx.x >> 5;
  const int l15  = lane & 15;
  const int half = lane >> 4;
  const int m0   = blockIdx.y << 5;
  const int n0   = (blockIdx.x << 7) + (wave << 5);
  // NOTE: waves with n0 >= N (only in small-N layers) still compute garbage
  // tiles; their W reads stay inside the weight workspace region and their
  // results are discarded by the guarded epilogue. Keeps EXEC all-ones and
  // every thread at every barrier.

  const int srow = threadIdx.x >> 2;         // staging: row 0..31
  const int skc  = (threadIdx.x & 3) << 3;   // staging: k-chunk 0/8/16/24
  const bf16_t* Ast = A + (size_t)(m0 + srow) * K + skc;
  const bf16_t* Wp0 = W + (size_t)(n0 + l15) * K + half * 16;
  const bf16_t* Wp1 = Wp0 + (size_t)16 * K;

  v8f acc00 = {}, acc01 = {}, acc10 = {}, acc11 = {};

  async_stage_b128(&As[0][srow][skc], Ast);
  wait_async0();
  __syncthreads();

  for (int kt = 0; kt < K; kt += 32) {
    const int buf = (kt >> 5) & 1;
    if (kt + 32 < K)                          // async-stage next A panel
      async_stage_b128(&As[buf ^ 1][srow][skc], Ast + kt + 32);
    if (kt + 512 < K) {
      __builtin_prefetch(Wp0 + kt + 512, 0, 1);
      __builtin_prefetch(Wp1 + kt + 512, 0, 1);
    }
    v16bf a0 = frag_lds(&As[buf][l15][0], half);
    v16bf a1 = frag_lds(&As[buf][16 + l15][0], half);
    v16bf b0 = *(const v16bf*)(Wp0 + kt);
    v16bf b1 = *(const v16bf*)(Wp1 + kt);
    acc00 = __builtin_amdgcn_wmma_f32_16x16x32_bf16(false, a0, false, b0, (short)0, acc00, false, false);
    acc01 = __builtin_amdgcn_wmma_f32_16x16x32_bf16(false, a0, false, b1, (short)0, acc01, false, false);
    acc10 = __builtin_amdgcn_wmma_f32_16x16x32_bf16(false, a1, false, b0, (short)0, acc10, false, false);
    acc11 = __builtin_amdgcn_wmma_f32_16x16x32_bf16(false, a1, false, b1, (short)0, acc11, false, false);
    wait_async0();
    __syncthreads();
  }

  if (n0 >= N) return;
  const int c0  = n0 + l15;
  const int c1c = n0 + 16 + l15;
  float badd0 = (bias1 ? bias1[c0] : 0.f) + (bias2 ? bias2[c0] : 0.f);
  float badd1 = (bias1 ? bias1[c1c] : 0.f) + (bias2 ? bias2[c1c] : 0.f);
  const int r0 = m0 + (half << 3);
  store_tile(C, Cb, N, r0,      c0,  acc00, badd0, act);
  store_tile(C, Cb, N, r0,      c1c, acc01, badd1, act);
  store_tile(C, Cb, N, r0 + 16, c0,  acc10, badd0, act);
  store_tile(C, Cb, N, r0 + 16, c1c, acc11, badd1, act);
}

// ---------------------------------------------------------------------------
// Fused LSTM timestep: gates = h_prev @ Wh^T + Gin[t]; cell update; writes
// h (bf16) + output sequence slot. Block = 8 waves: wave = gate (0..3) x
// K-slice (0..1); split-K halves the serial WMMA chain. Partial gate tiles
// are summed in LDS by the elementwise phase. grid = H/16, block = 256.
// Batch B = 16 = WMMA M dim. H/2 % 32 == 0 required (H = 512 or 1024).
// ---------------------------------------------------------------------------
__global__ __launch_bounds__(256)
void lstm_step_kernel(const bf16_t* __restrict__ hprev, const bf16_t* __restrict__ Wh,
                      const float* __restrict__ Gin, float* __restrict__ cst,
                      bf16_t* __restrict__ hnext, bf16_t* __restrict__ out,
                      int H, int T, int t_in, int t_out, int Fstride, int colOff)
{
  __shared__ float g[8][16][16];
  const int lane = threadIdx.x & 31;
  const int wave = threadIdx.x >> 5;   // 0..7
  const int gate = wave & 3;           // 0=i 1=f 2=g 3=o
  const int ks   = wave >> 2;          // K-slice
  const int l15  = lane & 15;
  const int half = lane >> 4;
  const int n0   = blockIdx.x << 4;
  const int gcol = gate * H + n0;
  const int Hh   = H >> 1;
  const int kbeg = ks * Hh;

  const bf16_t* Ap = hprev + l15 * H + half * 16 + kbeg;
  const bf16_t* Wp = Wh + (size_t)(gcol + l15) * H + half * 16 + kbeg;

  v8f acc = {};
#pragma unroll 2
  for (int k = 0; k < Hh; k += 32) {
    v16bf a = load_a_frag(Ap, k);
    v16bf b = *(const v16bf*)(Wp + k);
    acc = __builtin_amdgcn_wmma_f32_16x16x32_bf16(false, a, false, b, (short)0, acc, false, false);
  }

  const int rbase = half << 3;
#pragma unroll
  for (int v = 0; v < 8; ++v)
    g[wave][rbase + v][l15] = acc[v];
  __syncthreads();

  // one (batch, col) element per thread
  const int b = threadIdx.x >> 4;
  const int j = threadIdx.x & 15;
  const long grow = ((long)b * T + t_in) * (4L * H) + n0 + j;
  float gi = g[0][b][j] + g[4][b][j] + Gin[grow];
  float gf = g[1][b][j] + g[5][b][j] + Gin[grow + H];
  float gg = g[2][b][j] + g[6][b][j] + Gin[grow + 2 * H];
  float go = g[3][b][j] + g[7][b][j] + Gin[grow + 3 * H];
  const int idx = b * H + n0 + j;
  float cc = sigm(gf) * cst[idx] + sigm(gi) * tanhf(gg);
  cst[idx] = cc;
  float hv = sigm(go) * tanhf(cc);
  hnext[idx] = (bf16_t)hv;
  out[((long)b * T + t_out) * Fstride + colOff + n0 + j] = (bf16_t)hv;
}

// ---------------------------------------------------------------------------
// f32 -> bf16 convert with optional row tail padding (weights: pad K to mult 32)
// ---------------------------------------------------------------------------
__global__ __launch_bounds__(256)
void cvt_bf16_kernel(const float* __restrict__ src, bf16_t* __restrict__ dst,
                     int rows, int Ksrc, int Kdst)
{
  long i = (long)blockIdx.x * blockDim.x + threadIdx.x;
  long total = (long)rows * Kdst;
  if (i >= total) return;
  int k = (int)(i % Kdst);
  long r = i / Kdst;
  dst[i] = (k < Ksrc) ? (bf16_t)src[r * Ksrc + k] : (bf16_t)0.f;
}

// ---------------------------------------------------------------------------
// Generic im2col (2D, also conv1d via KH=1), arbitrary input strides,
// writes bf16 patches [B*OH*OW, Kpad] with zero padding.
// ---------------------------------------------------------------------------
__global__ __launch_bounds__(256)
void im2col_kernel(const float* __restrict__ x, bf16_t* __restrict__ col,
                   int C, int H, int W,
                   long sb, long sc, long sh, long sw,
                   int OH, int OW, int KH, int KW,
                   int strh, int strw, int padh, int padw,
                   int Ksrc, int Kpad, long total)
{
  long i = (long)blockIdx.x * blockDim.x + threadIdx.x;
  if (i >= total) return;
  int k = (int)(i % Kpad);
  long r = i / Kpad;
  int ow = (int)(r % OW); r /= OW;
  int oh = (int)(r % OH);
  int b  = (int)(r / OH);
  float v = 0.f;
  if (k < Ksrc) {
    int kw = k % KW; int t = k / KW; int kh = t % KH; int c = t / KH;
    int ih = oh * strh - padh + kh;
    int iw = ow * strw - padw + kw;
    if (ih >= 0 && ih < H && iw >= 0 && iw < W)
      v = x[b * sb + c * sc + ih * sh + iw * sw];
  }
  col[i] = (bf16_t)v;
}

// 3x3 stride-2 VALID maxpool, channels-last [B,H,W,C] -> [B,OH,OW,C]
__global__ __launch_bounds__(256)
void maxpool_kernel(const float* __restrict__ x, float* __restrict__ y,
                    int B, int H, int W, int C, int OH, int OW)
{
  long i = (long)blockIdx.x * blockDim.x + threadIdx.x;
  long total = (long)B * OH * OW * C;
  if (i >= total) return;
  int c = (int)(i % C); long r = i / C;
  int ow = (int)(r % OW); r /= OW;
  int oh = (int)(r % OH);
  int b  = (int)(r / OH);
  float m = -3.0e38f;
#pragma unroll
  for (int kh = 0; kh < 3; ++kh)
#pragma unroll
    for (int kw = 0; kw < 3; ++kw) {
      int ih = oh * 2 + kh, iw = ow * 2 + kw;
      if (ih < H && iw < W)
        m = fmaxf(m, x[(((long)b * H + ih) * W + iw) * C + c]);
    }
  y[i] = m;
}

// y = relu(layernorm(x)*w+b) as bf16; one wave (32 lanes) per row
__global__ __launch_bounds__(256)
void ln_relu_kernel(const float* __restrict__ x, const float* __restrict__ w,
                    const float* __restrict__ b, bf16_t* __restrict__ y, int R, int D)
{
  int row = blockIdx.x * 8 + (threadIdx.x >> 5);
  if (row >= R) return;
  int lane = threadIdx.x & 31;
  const float* xr = x + (long)row * D;
  float s = 0.f, s2 = 0.f;
  for (int i = lane; i < D; i += 32) { float v = xr[i]; s += v; s2 += v * v; }
#pragma unroll
  for (int off = 16; off; off >>= 1) {
    s  += __shfl_xor(s,  off, 32);
    s2 += __shfl_xor(s2, off, 32);
  }
  float m = s / D;
  float var = s2 / D - m * m;
  float inv = rsqrtf(var + 1e-5f);
  bf16_t* yr = y + (long)row * D;
  for (int i = lane; i < D; i += 32) {
    float v = (xr[i] - m) * inv * w[i] + b[i];
    yr[i] = (bf16_t)fmaxf(v, 0.f);
  }
}

// [B,T,F] -> [B,F,T] (bf16)
__global__ __launch_bounds__(256)
void transpose_btf_kernel(const bf16_t* __restrict__ src, bf16_t* __restrict__ dst,
                          int B, int T, int F)
{
  long i = (long)blockIdx.x * blockDim.x + threadIdx.x;
  long total = (long)B * T * F;
  if (i >= total) return;
  int f = (int)(i % F); long r = i / F;
  int t = (int)(r % T);
  int b = (int)(r / T);
  dst[((long)b * F + f) * T + t] = src[i];
}

// out[r] = leaky(dot(s1[r,:], w) + b), one wave per row
__global__ __launch_bounds__(256)
void fc2_dot_kernel(const float* __restrict__ s1, const float* __restrict__ w,
                    const float* __restrict__ bptr, float* __restrict__ out, int R, int K)
{
  int row = blockIdx.x * 8 + (threadIdx.x >> 5);
  if (row >= R) return;
  int lane = threadIdx.x & 31;
  const float* sr = s1 + (long)row * K;
  float s = 0.f;
  for (int i = lane; i < K; i += 32) s += sr[i] * w[i];
#pragma unroll
  for (int off = 16; off; off >>= 1) s += __shfl_xor(s, off, 32);
  if (lane == 0) {
    float v = s + bptr[0];
    out[row] = v > 0.f ? v : 0.01f * v;
  }
}

// sim[x,y] = max_i sum_d A[x,d] * F[y,i,d];  A:[16,2048]  F:[16,36,2048]
__global__ __launch_bounds__(256)
void sim_kernel(const float* __restrict__ A, const float* __restrict__ F,
                float* __restrict__ out)
{
  __shared__ float red[256];
  int x = blockIdx.x, y = blockIdx.y;
  float best = -3.0e38f;
  for (int i = 0; i < 36; ++i) {
    float s = 0.f;
    for (int d = threadIdx.x; d < 2048; d += 256)
      s += A[x * 2048 + d] * F[((long)y * 36 + i) * 2048 + d];
    red[threadIdx.x] = s;
    __syncthreads();
    for (int o = 128; o; o >>= 1) {
      if ((int)threadIdx.x < o) red[threadIdx.x] += red[threadIdx.x + o];
      __syncthreads();
    }
    if (threadIdx.x == 0) best = fmaxf(best, red[0]);
    __syncthreads();
  }
  if (threadIdx.x == 0) out[x * 16 + y] = best;
}

// ===========================================================================
extern "C" void kernel_launch(void* const* d_in, const int* in_sizes, int n_in,
                              void* d_out, int out_size, void* d_ws, size_t ws_size,
                              hipStream_t stream)
{
  (void)in_sizes; (void)n_in; (void)out_size; (void)ws_size;

  // ---- inputs (setup_inputs dict order, DFS through nested dicts) ----
  const float* audio  = (const float*)d_in[0];   // [16,40,256]
  const float* image  = (const float*)d_in[1];   // [16,3,224,224]
  const float* conv_w = (const float*)d_in[2];   // [512,40,5]
  const float *ln_w[5], *ln_b[5];
  for (int k = 0; k < 5; ++k) { ln_w[k] = (const float*)d_in[3 + 2*k]; ln_b[k] = (const float*)d_in[4 + 2*k]; }
  const float* fc_w[4];
  for (int k = 0; k < 4; ++k) fc_w[k] = (const float*)d_in[13 + k];
  const float* fc4_w = (const float*)d_in[17];
  const float* fc4_b = (const float*)d_in[18];
  const float *rWi[4], *rWh[4], *rbi[4], *rbh[4];
  for (int k = 0; k < 4; ++k) {
    rWi[k] = (const float*)d_in[19 + 4*k]; rWh[k] = (const float*)d_in[20 + 4*k];
    rbi[k] = (const float*)d_in[21 + 4*k]; rbh[k] = (const float*)d_in[22 + 4*k];
  }
  const float* e1[8]; for (int k = 0; k < 8; ++k) e1[k] = (const float*)d_in[35 + k];
  const float* e2[8]; for (int k = 0; k < 8; ++k) e2[k] = (const float*)d_in[43 + k];
  const float* afc1_w = (const float*)d_in[51];
  const float* afc1_b = (const float*)d_in[52];
  const float* afc2_w = (const float*)d_in[53];
  const float* afc2_b = (const float*)d_in[54];
  const float *cw[6], *cb[6];
  for (int k = 0; k < 6; ++k) { cw[k] = (const float*)d_in[55 + 2*k]; cb[k] = (const float*)d_in[56 + 2*k]; }

  // ---- workspace layout ----
  char* ws = (char*)d_ws;
  size_t off = 0;
  auto alloc = [&](size_t bytes) -> void* {
    off = (off + 255) & ~(size_t)255;
    void* p = ws + off; off += bytes; return p;
  };
  bf16_t* wgtA = (bf16_t*)alloc((size_t)4718592 * 2);   // largest weight (c6)
  bf16_t* wgtB = (bf16_t*)alloc((size_t)4194304 * 2);   // largest Wh (ernn2)
  bf16_t* col  = (bf16_t*)alloc((size_t)18688000 * 2);  // largest im2col (c2, Mpad)
  float*  act0 = (float*) alloc((size_t)4194304 * 4);   // conv acts / s1
  float*  act1 = (float*) alloc((size_t)1048576 * 4);   // pooled acts
  float*  feat = (float*) alloc((size_t)1179648 * 4);   // image features [16,36,2048]
  float*  zf32 = (float*) alloc((size_t)2097152 * 4);   // audio fc-stack f32
  bf16_t* zbf  = (bf16_t*)alloc((size_t)2097152 * 2);   // ln_relu output
  bf16_t* xbuf = (bf16_t*)alloc((size_t)8388608 * 2);   // layer IO ping
  bf16_t* ybuf = (bf16_t*)alloc((size_t)8388608 * 2);   // layer IO pong
  float*  gin  = (float*) alloc((size_t)16777216 * 4);  // gate precompute [4096,4H]
  bf16_t* h0   = (bf16_t*)alloc((size_t)16384 * 2);
  bf16_t* h1   = (bf16_t*)alloc((size_t)16384 * 2);
  float*  cst  = (float*) alloc((size_t)16384 * 4);
  float*  avec = (float*) alloc((size_t)32768 * 4);     // final audio vec [16,2048]
  bf16_t* st   = col;                                   // reuse: [B,2048,256] bf16

  // ---- host helpers ----
  auto cvt = [&](const float* src, bf16_t* dst, int rows, int Ksrc, int Kdst) {
    long total = (long)rows * Kdst;
    cvt_bf16_kernel<<<dim3((unsigned)((total + 255) / 256)), dim3(256), 0, stream>>>(
        src, dst, rows, Ksrc, Kdst);
  };
  auto gemm = [&](const bf16_t* A, const bf16_t* Wt, const float* b1, const float* b2,
                  float* C, bf16_t* Cb, int M, int N, int K, int act) {
    int Mpad = (M + 31) & ~31;
    dim3 g((unsigned)((N + 127) / 128), (unsigned)(Mpad / 32));
    gemm_bf16_wmma<<<g, dim3(128), 0, stream>>>(A, Wt, b1, b2, C, Cb, N, K, act);
  };
  auto im2col = [&](const float* x, int B, int C, int H, int W,
                    long sb, long sc, long sh, long sw,
                    int OH, int OW, int KH, int KW, int strh, int strw,
                    int padh, int padw, int Kpad) {
    int Ksrc = C * KH * KW;
    long total = (long)B * OH * OW * Kpad;
    im2col_kernel<<<dim3((unsigned)((total + 255) / 256)), dim3(256), 0, stream>>>(
        x, col, C, H, W, sb, sc, sh, sw, OH, OW, KH, KW,
        strh, strw, padh, padw, Ksrc, Kpad, total);
  };
  auto pool = [&](const float* x, float* y, int B, int H, int W, int C, int OH, int OW) {
    long total = (long)B * OH * OW * C;
    maxpool_kernel<<<dim3((unsigned)((total + 255) / 256)), dim3(256), 0, stream>>>(
        x, y, B, H, W, C, OH, OW);
  };
  auto lstm_dir = [&](const bf16_t* X, int Din, int H,
                      const float* Wi, const float* Wh, const float* bi, const float* bh,
                      bf16_t* out, int Fstride, int colOff, bool backward) {
    cvt(Wi, wgtA, 4 * H, Din, Din);
    gemm(X, wgtA, bi, bh, gin, nullptr, 4096, 4 * H, Din, 0);   // Gin for all (b,t)
    cvt(Wh, wgtB, 4 * H, H, H);
    hipMemsetAsync(h0, 0, (size_t)16 * 1024 * sizeof(bf16_t), stream);
    hipMemsetAsync(h1, 0, (size_t)16 * 1024 * sizeof(bf16_t), stream);
    hipMemsetAsync(cst, 0, (size_t)16 * 1024 * sizeof(float), stream);
    for (int s = 0; s < 256; ++s) {
      int t = backward ? (255 - s) : s;
      bf16_t* hp = (s & 1) ? h1 : h0;
      bf16_t* hn = (s & 1) ? h0 : h1;
      lstm_step_kernel<<<dim3((unsigned)(H / 16)), dim3(256), 0, stream>>>(
          hp, wgtB, gin, cst, hn, out, H, 256, t, t, Fstride, colOff);
    }
  };

  // ======================= IMAGE PATH =======================
  // c1: [16,3,224,224] NCHW, k11 s4 p2 -> [16,55,55,64]
  im2col(image, 16, 3, 224, 224, 3L*224*224, 224L*224, 224, 1,
         55, 55, 11, 11, 4, 4, 2, 2, 384);
  cvt(cw[0], wgtA, 64, 363, 384);
  gemm(col, wgtA, cb[0], nullptr, act0, nullptr, 48400, 64, 384, 1);
  pool(act0, act1, 16, 55, 55, 64, 27, 27);
  // c2: k5 s1 p2, C=64 -> 192
  im2col(act1, 16, 64, 27, 27, 27L*27*64, 1, 27L*64, 64,
         27, 27, 5, 5, 1, 1, 2, 2, 1600);
  cvt(cw[1], wgtA, 192, 1600, 1600);
  gemm(col, wgtA, cb[1], nullptr, act0, nullptr, 11664, 192, 1600, 1);
  pool(act0, act1, 16, 27, 27, 192, 13, 13);
  // c3: k3 p1, 192 -> 384
  im2col(act1, 16, 192, 13, 13, 13L*13*192, 1, 13L*192, 192,
         13, 13, 3, 3, 1, 1, 1, 1, 1728);
  cvt(cw[2], wgtA, 384, 1728, 1728);
  gemm(col, wgtA, cb[2], nullptr, act0, nullptr, 2704, 384, 1728, 1);
  // c4: 384 -> 256
  im2col(act0, 16, 384, 13, 13, 13L*13*384, 1, 13L*384, 384,
         13, 13, 3, 3, 1, 1, 1, 1, 3456);
  cvt(cw[3], wgtA, 256, 3456, 3456);
  gemm(col, wgtA, cb[3], nullptr, act1, nullptr, 2704, 256, 3456, 1);
  // c5: 256 -> 256
  im2col(act1, 16, 256, 13, 13, 13L*13*256, 1, 13L*256, 256,
         13, 13, 3, 3, 1, 1, 1, 1, 2304);
  cvt(cw[4], wgtA, 256, 2304, 2304);
  gemm(col, wgtA, cb[4], nullptr, act0, nullptr, 2704, 256, 2304, 1);
  pool(act0, act1, 16, 13, 13, 256, 6, 6);
  // c6: 256 -> 2048, no relu -> feat [16,36,2048]
  im2col(act1, 16, 256, 6, 6, 6L*6*256, 1, 6L*256, 256,
         6, 6, 3, 3, 1, 1, 1, 1, 2304);
  cvt(cw[5], wgtA, 2048, 2304, 2304);
  gemm(col, wgtA, cb[5], nullptr, feat, nullptr, 576, 2048, 2304, 0);

  // ======================= AUDIO PATH =======================
  // conv1d: [16,40,256], k5 p2 -> relu -> [16,256,512] (= [4096,512])
  im2col(audio, 16, 40, 1, 256, 40L*256, 256, 0, 1,
         1, 256, 1, 5, 1, 1, 0, 2, 224);
  cvt(conv_w, wgtA, 512, 200, 224);
  gemm(col, wgtA, nullptr, nullptr, zf32, nullptr, 4096, 512, 224, 1);

  // encoder FC stack: z = relu(LN(z)) @ fcK^T
  for (int k = 0; k < 4; ++k) {
    ln_relu_kernel<<<dim3(512), dim3(256), 0, stream>>>(zf32, ln_w[k], ln_b[k], zbf, 4096, 512);
    cvt(fc_w[k], wgtA, 512, 512, 512);
    gemm(zbf, wgtA, nullptr, nullptr, zf32, nullptr, 4096, 512, 512, 0);
  }
  ln_relu_kernel<<<dim3(512), dim3(256), 0, stream>>>(zf32, ln_w[4], ln_b[4], zbf, 4096, 512);
  cvt(fc4_w, wgtA, 64, 512, 512);
  gemm(zbf, wgtA, fc4_b, nullptr, zf32, xbuf, 4096, 64, 512, 0);  // bf16 out -> xbuf

  // 4 unidirectional LSTM layers
  bf16_t* X = xbuf; bf16_t* Y = ybuf;
  int din = 64;
  for (int k = 0; k < 4; ++k) {
    lstm_dir(X, din, 512, rWi[k], rWh[k], rbi[k], rbh[k], Y, 512, 0, false);
    { bf16_t* t = X; X = Y; Y = t; }
    din = 512;
  }
  // ernn1: bi-LSTM H=512 -> [B,T,1024]
  lstm_dir(X, 512, 512, e1[0], e1[1], e1[2], e1[3], Y, 1024, 0,   false);
  lstm_dir(X, 512, 512, e1[4], e1[5], e1[6], e1[7], Y, 1024, 512, true);
  { bf16_t* t = X; X = Y; Y = t; }
  // ernn2: bi-LSTM H=1024 -> [B,T,2048]
  lstm_dir(X, 1024, 1024, e2[0], e2[1], e2[2], e2[3], Y, 2048, 0,    false);
  lstm_dir(X, 1024, 1024, e2[4], e2[5], e2[6], e2[7], Y, 2048, 1024, true);
  { bf16_t* t = X; X = Y; Y = t; }

  // s: [B,T,2048] -> [B,2048,T] bf16, then fc1 (leaky) and fc2 (dot, leaky)
  transpose_btf_kernel<<<dim3(32768), dim3(256), 0, stream>>>(X, st, 16, 256, 2048);
  cvt(afc1_w, wgtA, 128, 256, 256);
  gemm(st, wgtA, afc1_b, nullptr, act0, nullptr, 32768, 128, 256, 2);
  fc2_dot_kernel<<<dim3(4096), dim3(256), 0, stream>>>(act0, afc2_w, afc2_b, avec, 32768, 128);

  // ======================= SIMILARITY =======================
  sim_kernel<<<dim3(16, 16), dim3(256), 0, stream>>>(avec, feat, (float*)d_out);
}
